// MSDeformAttn_64287070487203
// MI455X (gfx1250) — matrix operand order repaired and checked by reading, not compiled, not run on hardware
//
#include <hip/hip_runtime.h>
#include <hip/hip_bf16.h>

// ---- problem constants (fixed by the reference) ----
#define C_MODEL   256
#define N_HEADS   8
#define N_LEVELS  4
#define N_POINTS  4
#define HEAD_DIM  32
#define BATCH     4
#define LQ        5440          // 64*64 + 32*32 + 16*16 + 8*8
#define LV        5440
#define MROWS     (BATCH * LQ)  // 21760, divisible by 16
#define NOFF      256           // nH*nL*nP*2
#define NATT      128           // nH*nL*nP
#define NOA       384           // NOFF + NATT

typedef __attribute__((ext_vector_type(16))) _Float16 v16h;
typedef __attribute__((ext_vector_type(8)))  _Float16 v8h;
typedef __attribute__((ext_vector_type(8)))  float    v8f;

// ---------- fp32 -> fp16 elementwise ----------
__global__ void cvt_f16_kernel(const float* __restrict__ s, _Float16* __restrict__ d, int n) {
    int i = blockIdx.x * blockDim.x + threadIdx.x;
    if (i < n) d[i] = (_Float16)s[i];
}

// ---------- fp32 [K][N] -> fp16 [N][K] (weight transpose for WMMA B frags) ----------
__global__ void transpose_f16_kernel(const float* __restrict__ w, _Float16* __restrict__ wt,
                                     int K, int N) {
    int i = blockIdx.x * blockDim.x + threadIdx.x;
    if (i < K * N) {
        int k = i / N, n = i - k * N;
        wt[(size_t)n * K + k] = (_Float16)w[i];
    }
}

// ---------- concat [b_off | b_attn] ----------
__global__ void pack_bias_kernel(const float* __restrict__ b_off,
                                 const float* __restrict__ b_attn,
                                 float* __restrict__ boa) {
    int i = threadIdx.x;
    if (i < NOFF)      boa[i] = b_off[i];
    else if (i < NOA)  boa[i] = b_attn[i - NOFF];
}

// ---------- WMMA GEMM: C[M][N] = A_f16[M][K] * Wt_f16[N][K] + bias[N] ----------
// One wave computes one 16x16 f32 tile via K/32 steps of v_wmma_f32_16x16x32_f16.
// Fragment layouts (A/B: contiguous k order, same in both; C/D per CDNA5 ISA 7.12.2):
//   A 16x32 f16 : lane = m + 16*g, v16h[e]   = K (16g + e)
//   B 32x16 f16 : lane = n + 16*g, v16h[e]   = K (16g + e)
//   C/D 16x16 f32: v8f[r] -> row r + 8*(lane>>4), col lane&15
__global__ void wmma_gemm_bias_kernel(const _Float16* __restrict__ A,
                                      const _Float16* __restrict__ Wt,
                                      const float* __restrict__ bias,
                                      float* __restrict__ C,
                                      int M, int N, int K) {
    const int wave = (int)((blockIdx.x * blockDim.x + threadIdx.x) >> 5);
    const int lane = threadIdx.x & 31;
    const int nt = N >> 4;
    const int tiles = (M >> 4) * nt;
    if (wave >= tiles) return;                 // wave-uniform: EXEC stays all-ones
    const int tm = wave / nt, tn = wave - tm * nt;
    const int m0 = tm << 4,  n0 = tn << 4;
    const int r15 = lane & 15;
    const int g   = lane >> 4;

    const _Float16* pa = A  + (size_t)(m0 + r15) * K + (g << 4);   // +16*g halves
    const _Float16* pb = Wt + (size_t)(n0 + r15) * K + (g << 4);   // +16*g halves

    v8f acc = {};
    for (int kb = 0; kb < K; kb += 32) {
        v8h alo = *(const v8h*)(pa + kb);        // K = kb + 16g + (0..7)
        v8h ahi = *(const v8h*)(pa + kb + 8);    // K = kb + 16g + 8 + (0..7)
        v16h a;
#pragma unroll
        for (int i = 0; i < 8; ++i) { a[i] = alo[i]; a[i + 8] = ahi[i]; }
        v16h b = *(const v16h*)(pb + kb);        // K = kb + 16g + (0..15)
        acc = __builtin_amdgcn_wmma_f32_16x16x32_f16(
                  false, a, false, b, (short)0, acc, false, false);
    }

    const float bv = bias[n0 + r15];
    float* pc = C + (size_t)(m0 + (g << 3)) * N + (n0 + r15);
#pragma unroll
    for (int r = 0; r < 8; ++r) pc[(size_t)r * N] = acc[r] + bv;
}

// ---------- deformable sampling core ----------
// grid.x = b*LQ + q ; block = 256 threads = 8 waves, one wave per head.
// lane = channel (coalesced 128B gathers per corner); addressing math is wave-uniform.
__global__ void msda_sample_kernel(const float* __restrict__ oa,    // [M][384]
                                   const float* __restrict__ refp,  // [B][LQ][4][2]
                                   const float* __restrict__ val,   // [B][LV][256] = [.,.,nH,d]
                                   _Float16* __restrict__ acc_h) {  // [M][256]
    const int m = blockIdx.x;
    const int h = threadIdx.x >> 5;
    const int c = threadIdx.x & 31;
    const int b = m / LQ;

    const float* oarow = oa + (size_t)m * NOA;

    // softmax over the 16 (level, point) logits of this head
    const float* logit = oarow + NOFF + h * (N_LEVELS * N_POINTS);
    float lg[16];
    float mx = -1e30f;
#pragma unroll
    for (int j = 0; j < 16; ++j) { lg[j] = logit[j]; mx = fmaxf(mx, lg[j]); }
    float den = 0.f;
#pragma unroll
    for (int j = 0; j < 16; ++j) { lg[j] = __expf(lg[j] - mx); den += lg[j]; }
    const float inv = 1.f / den;

    const int   DIM[N_LEVELS]   = {64, 32, 16, 8};
    const int   START[N_LEVELS] = {0, 4096, 5120, 5376};

    const float* ref   = refp + (size_t)m * (N_LEVELS * 2);
    const float* voff  = oarow + h * (N_LEVELS * N_POINTS * 2);
    const float* vbase = val + (size_t)b * LV * C_MODEL + h * HEAD_DIM + c;

    float sum = 0.f;
#pragma unroll
    for (int l = 0; l < N_LEVELS; ++l) {
        const int   W  = DIM[l], H = DIM[l];
        const float fw = (float)W, fh = (float)H;
        const float rx = ref[l * 2 + 0], ry = ref[l * 2 + 1];
        const float* vl = vbase + (size_t)START[l] * C_MODEL;
#pragma unroll
        for (int p = 0; p < N_POINTS; ++p) {
            const float ox = voff[(l * N_POINTS + p) * 2 + 0];
            const float oy = voff[(l * N_POINTS + p) * 2 + 1];
            // loc = ref + off/(W,H); grid coords: loc*W - 0.5
            const float x = (rx + ox / fw) * fw - 0.5f;
            const float y = (ry + oy / fh) * fh - 0.5f;
            const float x0f = floorf(x), y0f = floorf(y);
            const float wx1 = x - x0f,  wy1 = y - y0f;
            const int x0 = (int)x0f, y0 = (int)y0f;
            const float aw = lg[l * N_POINTS + p] * inv;
            float g = 0.f;
#pragma unroll
            for (int dy = 0; dy < 2; ++dy) {
                const int yi = y0 + dy;
                if (yi < 0 || yi >= H) continue;
                const float wy = dy ? wy1 : 1.f - wy1;
#pragma unroll
                for (int dx = 0; dx < 2; ++dx) {
                    const int xi = x0 + dx;
                    if (xi < 0 || xi >= W) continue;
                    const float wx = dx ? wx1 : 1.f - wx1;
                    g += wy * wx * vl[(size_t)(yi * W + xi) * C_MODEL];
                }
            }
            sum += aw * g;
        }
    }
    acc_h[(size_t)m * C_MODEL + h * HEAD_DIM + c] = (_Float16)sum;
}

extern "C" void kernel_launch(void* const* d_in, const int* in_sizes, int n_in,
                              void* d_out, int out_size, void* d_ws, size_t ws_size,
                              hipStream_t stream) {
    (void)in_sizes; (void)n_in; (void)out_size; (void)ws_size;

    const float* query  = (const float*)d_in[0];
    const float* refp   = (const float*)d_in[1];
    const float* xin    = (const float*)d_in[2];
    // d_in[3] spatial_shapes / d_in[4] level_start_index are compile-time constants here
    const float* w_val  = (const float*)d_in[5];
    const float* b_val  = (const float*)d_in[6];
    const float* w_off  = (const float*)d_in[7];
    const float* b_off  = (const float*)d_in[8];
    const float* w_attn = (const float*)d_in[9];
    const float* b_attn = (const float*)d_in[10];
    const float* w_out  = (const float*)d_in[11];
    const float* b_out  = (const float*)d_in[12];
    float* out = (float*)d_out;

    char*  ws  = (char*)d_ws;
    size_t off = 0;
    auto take = [&](size_t bytes) -> char* {
        char* p = ws + off;
        off = (off + bytes + 255) & ~(size_t)255;
        return p;
    };
    _Float16* xh   = (_Float16*)take((size_t)MROWS * C_MODEL * 2);
    _Float16* qh   = (_Float16*)take((size_t)MROWS * C_MODEL * 2);
    _Float16* wvT  = (_Float16*)take((size_t)C_MODEL * C_MODEL * 2);
    _Float16* woaT = (_Float16*)take((size_t)NOA * C_MODEL * 2);
    _Float16* woT  = (_Float16*)take((size_t)C_MODEL * C_MODEL * 2);
    float*    boa  = (float*)take(NOA * 4);
    float*    valb = (float*)take((size_t)MROWS * C_MODEL * 4);
    float*    oab  = (float*)take((size_t)MROWS * NOA * 4);
    _Float16* acch = xh;  // xh is dead after the value GEMM; reuse (stream-ordered)

    const int TPB = 256;
    const int nelem = MROWS * C_MODEL;

    // activations -> f16
    cvt_f16_kernel<<<(nelem + TPB - 1) / TPB, TPB, 0, stream>>>(xin, xh, nelem);
    cvt_f16_kernel<<<(nelem + TPB - 1) / TPB, TPB, 0, stream>>>(query, qh, nelem);

    // weights -> f16, transposed to [N][K]
    transpose_f16_kernel<<<(C_MODEL * C_MODEL + TPB - 1) / TPB, TPB, 0, stream>>>(
        w_val, wvT, C_MODEL, C_MODEL);
    transpose_f16_kernel<<<(C_MODEL * NOFF + TPB - 1) / TPB, TPB, 0, stream>>>(
        w_off, woaT, C_MODEL, NOFF);
    transpose_f16_kernel<<<(C_MODEL * NATT + TPB - 1) / TPB, TPB, 0, stream>>>(
        w_attn, woaT + (size_t)NOFF * C_MODEL, C_MODEL, NATT);
    transpose_f16_kernel<<<(C_MODEL * C_MODEL + TPB - 1) / TPB, TPB, 0, stream>>>(
        w_out, woT, C_MODEL, C_MODEL);
    pack_bias_kernel<<<1, NOA, 0, stream>>>(b_off, b_attn, boa);

    // value = input_flatten @ w_val + b_val
    {
        int waves = (MROWS / 16) * (C_MODEL / 16);
        wmma_gemm_bias_kernel<<<(waves + 7) / 8, 256, 0, stream>>>(
            xh, wvT, b_val, valb, MROWS, C_MODEL, C_MODEL);
    }
    // [off | attn_logits] = query @ [w_off | w_attn] + [b_off | b_attn]
    {
        int waves = (MROWS / 16) * (NOA / 16);
        wmma_gemm_bias_kernel<<<(waves + 7) / 8, 256, 0, stream>>>(
            qh, woaT, boa, oab, MROWS, NOA, C_MODEL);
    }
    // softmax + bilinear sampling + per-head accumulation
    msda_sample_kernel<<<MROWS, 256, 0, stream>>>(oab, refp, valb, acch);
    // result = acc @ w_out + b_out
    {
        int waves = (MROWS / 16) * (C_MODEL / 16);
        wmma_gemm_bias_kernel<<<(waves + 7) / 8, 256, 0, stream>>>(
            acch, woT, b_out, out, MROWS, C_MODEL, C_MODEL);
    }
}
